// SumLayer_4501125726250
// MI455X (gfx1250) — matrix operand order repaired and checked by reading, not compiled, not run on hardware
//
#include <hip/hip_runtime.h>

typedef _Float16 half8  __attribute__((ext_vector_type(8)));
typedef _Float16 half4  __attribute__((ext_vector_type(4)));
typedef _Float16 v16h   __attribute__((ext_vector_type(16)));
typedef float    v8f    __attribute__((ext_vector_type(8)));

#define BATCH 128
#define NREG  256
#define NNODE 512
#define NSUM  512

// ---------------------------------------------------------------------------
// Kernel 1: row softmax of weight (r*s rows of 512 f32) -> f16, row-major.
// One wave per row (wave32), 16 elems/lane as 4x float4 -> 4x half4.
// ---------------------------------------------------------------------------
__global__ __launch_bounds__(256) void softmax_rows_f16(
    const float* __restrict__ w, _Float16* __restrict__ p) {
  const int lane = threadIdx.x & 31;
  const int wave = threadIdx.x >> 5;
  const long long row = (long long)blockIdx.x * 8 + wave;
  const float* wr = w + row * NNODE;
  _Float16* pr = p + row * NNODE;

  float4 v[4];
  float mx = -3.402823466e38f;
#pragma unroll
  for (int i = 0; i < 4; ++i) {
    v[i] = ((const float4*)wr)[i * 32 + lane];
    mx = fmaxf(mx, fmaxf(fmaxf(v[i].x, v[i].y), fmaxf(v[i].z, v[i].w)));
  }
#pragma unroll
  for (int off = 16; off; off >>= 1) mx = fmaxf(mx, __shfl_xor(mx, off, 32));

  float sum = 0.f;
#pragma unroll
  for (int i = 0; i < 4; ++i) {
    v[i].x = __expf(v[i].x - mx);
    v[i].y = __expf(v[i].y - mx);
    v[i].z = __expf(v[i].z - mx);
    v[i].w = __expf(v[i].w - mx);
    sum += (v[i].x + v[i].y) + (v[i].z + v[i].w);
  }
#pragma unroll
  for (int off = 16; off; off >>= 1) sum += __shfl_xor(sum, off, 32);
  const float inv = 1.0f / sum;

#pragma unroll
  for (int i = 0; i < 4; ++i) {
    half4 h;
    h.x = (_Float16)(v[i].x * inv);
    h.y = (_Float16)(v[i].y * inv);
    h.z = (_Float16)(v[i].z * inv);
    h.w = (_Float16)(v[i].w * inv);
    ((half4*)pr)[i * 32 + lane] = h;
  }
}

// ---------------------------------------------------------------------------
// Kernel 2: per (b,r) row of x: m = max_n x, ex = exp(x - m) in f16.
// ---------------------------------------------------------------------------
__global__ __launch_bounds__(256) void rowmax_exp_f16(
    const float* __restrict__ x, _Float16* __restrict__ ex,
    float* __restrict__ m) {
  const int lane = threadIdx.x & 31;
  const int wave = threadIdx.x >> 5;
  const long long row = (long long)blockIdx.x * 8 + wave;
  const float* xr = x + row * NNODE;
  _Float16* er = ex + row * NNODE;

  float4 v[4];
  float mx = -3.402823466e38f;
#pragma unroll
  for (int i = 0; i < 4; ++i) {
    v[i] = ((const float4*)xr)[i * 32 + lane];
    mx = fmaxf(mx, fmaxf(fmaxf(v[i].x, v[i].y), fmaxf(v[i].z, v[i].w)));
  }
#pragma unroll
  for (int off = 16; off; off >>= 1) mx = fmaxf(mx, __shfl_xor(mx, off, 32));
  if (lane == 0) m[row] = mx;

#pragma unroll
  for (int i = 0; i < 4; ++i) {
    half4 h;
    h.x = (_Float16)__expf(v[i].x - mx);
    h.y = (_Float16)__expf(v[i].y - mx);
    h.z = (_Float16)__expf(v[i].z - mx);
    h.w = (_Float16)__expf(v[i].w - mx);
    ((half4*)er)[i * 32 + lane] = h;
  }
}

// ---------------------------------------------------------------------------
// Kernel 3: per-region WMMA GEMM C[b,s] = sum_n ex[b,n]*p[s,n], fused
// epilogue out = m[b,r] + log(C). One wave -> 16(b) x 64(s) strip; A fragment
// reused across 4 B tiles (4x v_wmma_f32_16x16x32_f16 per K step, 16 steps).
// Grid divides evenly: EXEC is all-ones for every WMMA (ISA requirement).
// ---------------------------------------------------------------------------
__global__ __launch_bounds__(256) void logsemiring_wmma_gemm(
    const _Float16* __restrict__ exh,  // (BATCH, NREG, NNODE) f16
    const _Float16* __restrict__ ph,   // (NREG, NSUM, NNODE) f16
    const float* __restrict__ m,       // (BATCH, NREG)
    float* __restrict__ out) {         // (BATCH, NREG, NSUM)
  const int lane = threadIdx.x & 31;
  const int wave = threadIdx.x >> 5;
  const int r  = blockIdx.x >> 3;
  const int jr = ((blockIdx.x & 7) << 3) | wave;  // 0..63 jobs per region
  const int b0 = (jr >> 3) * 16;                  // batch tile base
  const int s0 = (jr & 7) * 64;                   // s strip base (4 tiles)
  const int mrow = lane & 15;                     // row-in-tile
  const int hi   = lane >> 4;                     // lane half select

  // A: lane holds ex row b0+mrow, 16 consecutive K starting at hi*16
  const _Float16* ar =
      exh + ((long long)(b0 + mrow) * NREG + r) * NNODE + hi * 16;
  // B: lane holds p row s0+t*16+mrow, 16 consecutive n starting at hi*16
  const _Float16* pbase =
      ph + ((long long)r * NSUM + s0 + mrow) * NNODE + hi * 16;
  const _Float16* pr0 = pbase;
  const _Float16* pr1 = pbase + 16LL * NNODE;
  const _Float16* pr2 = pbase + 32LL * NNODE;
  const _Float16* pr3 = pbase + 48LL * NNODE;

  v8f acc[4] = {v8f{}, v8f{}, v8f{}, v8f{}};

  for (int k0 = 0; k0 < NNODE; k0 += 32) {
    half8 alo = *(const half8*)(ar);
    half8 ahi = *(const half8*)(ar + 8);
    v16h a;
#pragma unroll
    for (int t = 0; t < 8; ++t) { a[t] = alo[t]; a[8 + t] = ahi[t]; }

    v16h bv0 = *(const v16h*)(pr0);
    v16h bv1 = *(const v16h*)(pr1);
    v16h bv2 = *(const v16h*)(pr2);
    v16h bv3 = *(const v16h*)(pr3);

    acc[0] = __builtin_amdgcn_wmma_f32_16x16x32_f16(false, a, false, bv0,
                                                    (short)0, acc[0], false, false);
    acc[1] = __builtin_amdgcn_wmma_f32_16x16x32_f16(false, a, false, bv1,
                                                    (short)0, acc[1], false, false);
    acc[2] = __builtin_amdgcn_wmma_f32_16x16x32_f16(false, a, false, bv2,
                                                    (short)0, acc[2], false, false);
    acc[3] = __builtin_amdgcn_wmma_f32_16x16x32_f16(false, a, false, bv3,
                                                    (short)0, acc[3], false, false);

    ar += 32; pr0 += 32; pr1 += 32; pr2 += 32; pr3 += 32;
  }

  // Epilogue: D vgpr i -> b = b0 + i + 8*hi, N column = s0 + t*16 + (lane&15).
  const int scol = lane & 15;
#pragma unroll
  for (int i = 0; i < 8; ++i) {
    const int b = b0 + i + 8 * hi;
    const float mv = m[(long long)b * NREG + r];
    float* orow = out + ((long long)b * NREG + r) * NSUM + s0 + scol;
#pragma unroll
    for (int t = 0; t < 4; ++t) {
      orow[t * 16] = mv + logf(acc[t][i]);
    }
  }
}

// ---------------------------------------------------------------------------
extern "C" void kernel_launch(void* const* d_in, const int* in_sizes, int n_in,
                              void* d_out, int out_size, void* d_ws, size_t ws_size,
                              hipStream_t stream) {
  const float* x = (const float*)d_in[0];   // (128, 256, 512) f32
  const float* w = (const float*)d_in[1];   // (256, 512, 512) f32
  float* out = (float*)d_out;               // (128, 256, 512) f32

  char* ws = (char*)d_ws;
  const size_t p_bytes  = (size_t)NREG * NSUM * NNODE * sizeof(_Float16);   // 128 MB
  const size_t ex_bytes = (size_t)BATCH * NREG * NNODE * sizeof(_Float16);  // 32 MB
  _Float16* p  = (_Float16*)ws;
  _Float16* ex = (_Float16*)(ws + p_bytes);
  float*    m  = (float*)(ws + p_bytes + ex_bytes);

  // 1) softmax(weight) rows -> f16 (B-matrix-ready row-major layout)
  softmax_rows_f16<<<(NREG * NSUM) / 8, 256, 0, stream>>>(w, p);
  // 2) rowwise max + exp(x - m) -> f16
  rowmax_exp_f16<<<(BATCH * NREG) / 8, 256, 0, stream>>>(x, ex, m);
  // 3) per-region WMMA GEMM with fused m + log epilogue
  logsemiring_wmma_gemm<<<NREG * 8, 256, 0, stream>>>(ex, p, m, out);
}
